// Infidelity_51805895524432
// MI455X (gfx1250) — matrix operand order repaired and checked by reading, not compiled, not run on hardware
//
#include <hip/hip_runtime.h>

// ---------------------------------------------------------------------------
// Problem sizes (fixed by the reference)
// ---------------------------------------------------------------------------
#define BATCH   128
#define TLEN    3000
#define NCLS    5
#define NFREQ   50
#define HID     512
#define NROWS   (BATCH * NCLS)        // 640
#define NCOEF   11                    // order-5 bandstop -> 11 taps
#define PADL    33                    // 3 * NCOEF
#define EXTL    (TLEN + 2 * PADL)     // 3066
#define EXTSTR  3072                  // padded row stride for yext

typedef __attribute__((ext_vector_type(16))) _Float16 v16h;
typedef __attribute__((ext_vector_type(8)))  _Float16 v8h;
typedef __attribute__((ext_vector_type(8)))  float    v8f;

// ---------------------------------------------------------------------------
// fp64 complex helpers for the filter design (runs once, 50 threads)
// ---------------------------------------------------------------------------
struct C2 { double re, im; };
__device__ __forceinline__ C2 cadd(C2 a, C2 b) { return {a.re + b.re, a.im + b.im}; }
__device__ __forceinline__ C2 csub(C2 a, C2 b) { return {a.re - b.re, a.im - b.im}; }
__device__ __forceinline__ C2 cmul(C2 a, C2 b) {
    return {a.re * b.re - a.im * b.im, a.re * b.im + a.im * b.re};
}
__device__ __forceinline__ C2 cdiv(C2 a, C2 b) {
    double d = b.re * b.re + b.im * b.im;
    return {(a.re * b.re + a.im * b.im) / d, (a.im * b.re - a.re * b.im) / d};
}
__device__ __forceinline__ C2 csqrtc(C2 z) {
    double r = sqrt(z.re * z.re + z.im * z.im);
    double u = sqrt(0.5 * fmax(r + z.re, 0.0));
    double v = sqrt(0.5 * fmax(r - z.re, 0.0));
    if (z.im < 0.0) v = -v;
    return {u, v};
}

__device__ void poly_from_roots(const C2* r, int n, C2* c) {
    c[0] = {1.0, 0.0};
    for (int j = 1; j <= n; ++j) c[j] = {0.0, 0.0};
    for (int k = 0; k < n; ++k)
        for (int j = k + 1; j >= 1; --j)
            c[j] = csub(c[j], cmul(r[k], c[j - 1]));
}

// filters layout: per freq idx f (0..49), 32 floats: b[0..10], a[0..10], zi[0..9]
__global__ void design_kernel(float* __restrict__ filters) {
    int i = blockIdx.x * blockDim.x + threadIdx.x;
    if (i >= NFREQ) return;
    const double PI = 3.14159265358979323846;
    double freq = (double)i;
    double lo = (freq - 1.0 <= 0.0)  ? 0.1  : freq - 1.0;
    double hi = (freq + 1.0 >= 50.0) ? 49.9 : freq + 1.0;
    double w1p = 4.0 * tan(PI * (lo / 50.0) * 0.5);
    double w2p = 4.0 * tan(PI * (hi / 50.0) * 0.5);
    double wo = sqrt(w1p * w2p), bw = w2p - w1p;

    C2 pbs[10], zbs[10];
    C2 prodNegP = {1.0, 0.0};
    for (int k = 0; k < 5; ++k) {
        double th = PI * (double)(-4 + 2 * k) / 10.0;
        C2 p = {-cos(th), -sin(th)};                       // Butterworth prototype pole
        prodNegP = cmul(prodNegP, C2{-p.re, -p.im});
        C2 php = cdiv(C2{bw * 0.5, 0.0}, p);               // lp2bs
        C2 s   = csqrtc(csub(cmul(php, php), C2{wo * wo, 0.0}));
        pbs[k]     = cadd(php, s);
        pbs[k + 5] = csub(php, s);
        zbs[k]     = C2{0.0,  wo};
        zbs[k + 5] = C2{0.0, -wo};
    }
    double kbs = cdiv(C2{1.0, 0.0}, prodNegP).re;

    // bilinear, fs2 = 4
    C2 zd[10], pd[10], num = {1.0, 0.0}, den = {1.0, 0.0};
    for (int k = 0; k < 10; ++k) {
        C2 fz = csub(C2{4.0, 0.0}, zbs[k]);
        C2 fp = csub(C2{4.0, 0.0}, pbs[k]);
        num = cmul(num, fz); den = cmul(den, fp);
        zd[k] = cdiv(cadd(C2{4.0, 0.0}, zbs[k]), fz);
        pd[k] = cdiv(cadd(C2{4.0, 0.0}, pbs[k]), fp);
    }
    double kd = kbs * cdiv(num, den).re;

    C2 bcC[11], acC[11];
    poly_from_roots(zd, 10, bcC);
    poly_from_roots(pd, 10, acC);
    double bb[11], aa[11];
    for (int j = 0; j < 11; ++j) { bb[j] = kd * bcC[j].re; aa[j] = acC[j].re; }

    // lfilter_zi: solve (I - A^T) zi = b[1:] - a[1:]*b[0], A = companion
    double M[10][11];
    for (int r = 0; r < 10; ++r) {
        for (int c = 0; c < 10; ++c) {
            double at = (c == 0) ? -aa[r + 1] : ((r == c - 1) ? 1.0 : 0.0);
            M[r][c] = ((r == c) ? 1.0 : 0.0) - at;
        }
        M[r][10] = bb[r + 1] - aa[r + 1] * bb[0];
    }
    for (int col = 0; col < 10; ++col) {                   // partial pivoting
        int piv = col;
        for (int r = col + 1; r < 10; ++r)
            if (fabs(M[r][col]) > fabs(M[piv][col])) piv = r;
        if (piv != col)
            for (int c = col; c <= 10; ++c) { double t = M[col][c]; M[col][c] = M[piv][c]; M[piv][c] = t; }
        double d = M[col][col];
        for (int r = col + 1; r < 10; ++r) {
            double f = M[r][col] / d;
            for (int c = col; c <= 10; ++c) M[r][c] -= f * M[col][c];
        }
    }
    double zi[10];
    for (int r = 9; r >= 0; --r) {
        double s = M[r][10];
        for (int c = r + 1; c < 10; ++c) s -= M[r][c] * zi[c];
        zi[r] = s / M[r][r];
    }
    float* o = filters + i * 32;
    for (int j = 0; j < 11; ++j) { o[j] = (float)bb[j]; o[11 + j] = (float)aa[j]; }
    for (int j = 0; j < 10; ++j) o[22 + j] = (float)zi[j];
}

// ---------------------------------------------------------------------------
// Stable descending argsort of each W row (matches argsort(-W), stable ties)
// ---------------------------------------------------------------------------
__global__ void argsort_kernel(const float* __restrict__ W, int* __restrict__ imp) {
    int r = blockIdx.x * blockDim.x + threadIdx.x;
    if (r >= NROWS) return;
    const float* w = W + (size_t)r * NFREQ;
    float v[NFREQ]; bool used[NFREQ];
    for (int j = 0; j < NFREQ; ++j) { v[j] = w[j]; used[j] = false; }
    for (int f = 0; f < NFREQ; ++f) {
        int best = -1;
        for (int j = 0; j < NFREQ; ++j)
            if (!used[j] && (best < 0 || v[j] > v[best])) best = j;   // strict > => stable
        used[best] = true;
        imp[r * NFREQ + f] = best;
    }
}

// ---------------------------------------------------------------------------
// xf = broadcast(x) over classes; acc = 0
// ---------------------------------------------------------------------------
__global__ void init_kernel(const float* __restrict__ x, float* __restrict__ xf,
                            float* __restrict__ acc) {
    long i = (long)blockIdx.x * blockDim.x + threadIdx.x;
    if (i < (long)NROWS * TLEN) {
        int r = (int)(i / TLEN), t = (int)(i % TLEN);
        xf[i] = x[(size_t)(r / NCLS) * TLEN + t];
    }
    if (i < NROWS) acc[i] = 0.0f;
}

// ---------------------------------------------------------------------------
// filtfilt (odd padding, padlen 33) in-place on xf, one thread per row.
// Direct-form II transposed, order-10 state kept in registers.
// ---------------------------------------------------------------------------
__global__ void filt_kernel(float* __restrict__ xf, const float* __restrict__ filters,
                            const int* __restrict__ imp, int fstep,
                            float* __restrict__ yext) {
    int r = blockIdx.x * blockDim.x + threadIdx.x;
    if (r >= NROWS) return;
    float* x  = xf + (size_t)r * TLEN;
    float* ye = yext + (size_t)r * EXTSTR;
    const float* F = filters + imp[r * NFREQ + fstep] * 32;
    float bc[11], ac[11], zi[10], z[10];
#pragma unroll
    for (int j = 0; j < 11; ++j) { bc[j] = F[j]; ac[j] = F[11 + j]; }
#pragma unroll
    for (int j = 0; j < 10; ++j) zi[j] = F[22 + j];

    // ---- forward pass over the odd-extended signal ----
    float x0 = x[0], xL = x[TLEN - 1];
    float e0 = 2.0f * x0 - x[PADL];
#pragma unroll
    for (int j = 0; j < 10; ++j) z[j] = zi[j] * e0;
    for (int n = 0; n < EXTL; ++n) {
        float xn;
        if (n < PADL)             xn = 2.0f * x0 - x[PADL - n];
        else if (n < PADL + TLEN) xn = x[n - PADL];
        else                      xn = 2.0f * xL - x[2 * (PADL + TLEN) - PADL - 2 - n];
        float y = fmaf(bc[0], xn, z[0]);
#pragma unroll
        for (int j = 0; j < 9; ++j) z[j] = fmaf(bc[j + 1], xn, z[j + 1]) - ac[j + 1] * y;
        z[9] = bc[10] * xn - ac[10] * y;
        ye[n] = y;
    }
    // ---- backward pass; write the de-padded, re-reversed result into xf ----
    float yb0 = ye[EXTL - 1];
#pragma unroll
    for (int j = 0; j < 10; ++j) z[j] = zi[j] * yb0;
    for (int m = 0; m < EXTL; ++m) {
        float xn = ye[EXTL - 1 - m];
        float y = fmaf(bc[0], xn, z[0]);
#pragma unroll
        for (int j = 0; j < 9; ++j) z[j] = fmaf(bc[j + 1], xn, z[j + 1]) - ac[j + 1] * y;
        z[9] = bc[10] * xn - ac[10] * y;
        int o = (EXTL - 1 - PADL) - m;                 // 3032 - m
        if (o >= 0 && o < TLEN) x[o] = y;
    }
}

// ---------------------------------------------------------------------------
// H = relu(xf @ w1 + b1) via v_wmma_f32_16x16x32_f16.
// 256 threads (8 waves) -> 64x64 output tile; each wave -> 16x32 (2 WMMA/K-step).
// Double-buffered LDS panels; since TLEN % 8 == 0 every 8-wide staging group is
// all-in or all-out => uniform branches only, vector (b128) loads and stores.
// ---------------------------------------------------------------------------
__global__ __launch_bounds__(256)
void gemm_h_kernel(const float* __restrict__ A,   // [640 x 3000]
                   const float* __restrict__ w1,  // [3000 x 512]
                   const float* __restrict__ b1,  // [512]
                   float* __restrict__ H) {       // [640 x 512]
    __shared__ _Float16 sa[2][64][48];   // [buf][m][k]  stride 48 halfs = 96B (16B-aligned rows)
    __shared__ _Float16 sb[2][64][48];   // [buf][n][k]  K-transposed

    const int t = threadIdx.x;
    const int wv = t >> 5, lane = t & 31, laneM = lane & 15;
    const int m0 = blockIdx.x * 64;   // 10 M-blocks
    const int n0 = blockIdx.y * 64;   // 8  N-blocks
    const int ms  = wv & 3;           // M sub-tile of this wave
    const int nsb = (wv >> 2) * 2;    // first of two N sub-tiles

    const int aRow  = 16 * ms + laneM;
    const int koffA = (lane < 16) ? 0 : 16;   // contiguous K order: k = 16*(lane/16) + s
    const int bCol0 = 16 * nsb + laneM;
    const int bCol1 = bCol0 + 16;
    const int koffB = (lane < 16) ? 0 : 16;   // contiguous K order, same as A

    // cooperative loader coordinates
    const int la_m = t >> 2,  la_k = (t & 3) * 8;    // A: 64 rows x 32 K, 8 K per thread
    const int lb_n = t & 63,  lb_k = (t >> 6) * 8;   // B: 64 n x 32 K, 8 K per thread
    const float* arow = A  + (size_t)(m0 + la_m) * TLEN;
    const float* bcol = w1 + (size_t)(n0 + lb_n);

    auto stageA = [&](int k0, int buf) {
        v8h h;
        if (k0 + la_k + 7 < TLEN) {                 // whole group in-bounds (TLEN%8==0)
            float4 f0 = *(const float4*)(arow + k0 + la_k);
            float4 f1 = *(const float4*)(arow + k0 + la_k + 4);
            h[0] = (_Float16)f0.x; h[1] = (_Float16)f0.y;
            h[2] = (_Float16)f0.z; h[3] = (_Float16)f0.w;
            h[4] = (_Float16)f1.x; h[5] = (_Float16)f1.y;
            h[6] = (_Float16)f1.z; h[7] = (_Float16)f1.w;
        } else {
#pragma unroll
            for (int i = 0; i < 8; ++i) h[i] = (_Float16)0.0f;
        }
        *(v8h*)&sa[buf][la_m][la_k] = h;
    };
    auto stageB = [&](int k0, int buf) {
        v8h h;
        if (k0 + lb_k + 7 < TLEN) {                 // whole group in-bounds
#pragma unroll
            for (int i = 0; i < 8; ++i)
                h[i] = (_Float16)bcol[(size_t)(k0 + lb_k + i) * HID];
        } else {
#pragma unroll
            for (int i = 0; i < 8; ++i) h[i] = (_Float16)0.0f;
        }
        *(v8h*)&sb[buf][lb_n][lb_k] = h;
    };

    v8f c0 = {}, c1 = {};
    int p = 0;
    stageA(0, 0);
    stageB(0, 0);

    for (int k0 = 0; k0 < TLEN; k0 += 32) {
        __syncthreads();
        const int kn = k0 + 32;
        if (kn < TLEN) {                             // stage next panel into other buffer
            stageA(kn, p ^ 1);
            stageB(kn, p ^ 1);
            if (kn + 32 < TLEN) {                    // global_prefetch the panel after
                __builtin_prefetch(arow + kn + 32 + la_k, 0, 1);
                __builtin_prefetch(bcol + (size_t)(kn + 32 + lb_k) * HID, 0, 1);
            }
        }

        // A fragment: 16 consecutive K per lane {koffA..koffA+15}
        v8h alo = *(const v8h*)&sa[p][aRow][koffA];
        v8h ahi = *(const v8h*)&sa[p][aRow][koffA + 8];
        v16h av;
#pragma unroll
        for (int i = 0; i < 8; ++i) { av[i] = alo[i]; av[i + 8] = ahi[i]; }

        // B fragments: 16 consecutive K per lane (transposed LDS layout)
        v8h b0lo = *(const v8h*)&sb[p][bCol0][koffB];
        v8h b0hi = *(const v8h*)&sb[p][bCol0][koffB + 8];
        v8h b1lo = *(const v8h*)&sb[p][bCol1][koffB];
        v8h b1hi = *(const v8h*)&sb[p][bCol1][koffB + 8];
        v16h bv0, bv1;
#pragma unroll
        for (int i = 0; i < 8; ++i) {
            bv0[i] = b0lo[i]; bv0[i + 8] = b0hi[i];
            bv1[i] = b1lo[i]; bv1[i + 8] = b1hi[i];
        }

        c0 = __builtin_amdgcn_wmma_f32_16x16x32_f16(false, av, false, bv0,
                                                    (short)0, c0, false, false);
        c1 = __builtin_amdgcn_wmma_f32_16x16x32_f16(false, av, false, bv1,
                                                    (short)0, c1, false, false);
        p ^= 1;
    }

    // C/D layout: VGPR i -> M = i (+8 for lanes 16..31), N = lane%16
#pragma unroll
    for (int i = 0; i < 8; ++i) {
        int row = m0 + 16 * ms + i + ((lane < 16) ? 0 : 8);
        int col = n0 + 16 * nsb + laneM;
        H[(size_t)row * HID + col]      = fmaxf(c0[i] + b1[col],      0.0f);
        H[(size_t)row * HID + col + 16] = fmaxf(c1[i] + b1[col + 16], 0.0f);
    }
}

// ---------------------------------------------------------------------------
// acc[r] += weight * (H[r,:] . w2[:, r%5] + b2[r%5])  -- one wave32 per row
// ---------------------------------------------------------------------------
__global__ void accum_kernel(const float* __restrict__ H, const float* __restrict__ w2,
                             const float* __restrict__ b2, float* __restrict__ acc,
                             float weight) {
    int wv = threadIdx.x >> 5, lane = threadIdx.x & 31;
    int r = blockIdx.x * 8 + wv;
    if (r >= NROWS) return;
    int c = r % NCLS;
    const float* h = H + (size_t)r * HID;
    float s = 0.0f;
#pragma unroll
    for (int i = lane; i < HID; i += 32) s = fmaf(h[i], w2[i * NCLS + c], s);
#pragma unroll
    for (int off = 16; off > 0; off >>= 1) s += __shfl_down(s, off, 32);
    if (lane == 0) acc[r] += weight * (s + b2[c]);
}

// ---------------------------------------------------------------------------
// out = dx * (acc + 0.5 * model(zeros)), dx = 1/(F+2)
// ---------------------------------------------------------------------------
__global__ void finalize_kernel(const float* __restrict__ b1, const float* __restrict__ w2,
                                const float* __restrict__ b2, const float* __restrict__ acc,
                                float* __restrict__ out) {
    int r = blockIdx.x * blockDim.x + threadIdx.x;
    if (r >= NROWS) return;
    int c = r % NCLS;
    float s = 0.0f;
    for (int hh = 0; hh < HID; ++hh) s = fmaf(fmaxf(b1[hh], 0.0f), w2[hh * NCLS + c], s);
    s += b2[c];
    out[r] = (1.0f / 52.0f) * (acc[r] + 0.5f * s);
}

// ---------------------------------------------------------------------------
// host driver (graph-capture safe: launches only)
// ---------------------------------------------------------------------------
extern "C" void kernel_launch(void* const* d_in, const int* in_sizes, int n_in,
                              void* d_out, int out_size, void* d_ws, size_t ws_size,
                              hipStream_t stream) {
    const float* x   = (const float*)d_in[0];   // [128,3000]
    const float* W   = (const float*)d_in[1];   // [128,5,50]
    const float* w1p = (const float*)d_in[2];   // [3000,512]
    const float* b1p = (const float*)d_in[3];   // [512]
    const float* w2p = (const float*)d_in[4];   // [512,5]
    const float* b2p = (const float*)d_in[5];   // [5]
    float* out = (float*)d_out;                 // [128,5]

    char* ws = (char*)d_ws;
    size_t off = 0;
    float* filters = (float*)(ws + off); off += 8192;                          // 50*32 f32
    float* xf      = (float*)(ws + off); off += (size_t)NROWS * TLEN * 4;      // 7.68 MB
    float* Hbuf    = (float*)(ws + off); off += (size_t)NROWS * HID * 4;       // 1.31 MB
    float* yext    = (float*)(ws + off); off += (size_t)NROWS * EXTSTR * 4;    // 7.86 MB
    int*   imp     = (int*)  (ws + off); off += (size_t)NROWS * NFREQ * 4;     // 128 KB
    float* acc     = (float*)(ws + off); off += (size_t)NROWS * 4;
    (void)ws_size; (void)in_sizes; (void)n_in; (void)out_size;

    design_kernel <<<1, 64, 0, stream>>>(filters);
    argsort_kernel<<<(NROWS + 63) / 64, 64, 0, stream>>>(W, imp);

    long ninit = (long)NROWS * TLEN;
    init_kernel<<<(unsigned)((ninit + 255) / 256), 256, 0, stream>>>(x, xf, acc);

    dim3 gg(NROWS / 64, HID / 64);   // 10 x 8 blocks

    // trapezoid endpoint y[0] = model(x), weight 0.5
    gemm_h_kernel<<<gg, 256, 0, stream>>>(xf, w1p, b1p, Hbuf);
    accum_kernel<<<NROWS / 8, 256, 0, stream>>>(Hbuf, w2p, b2p, acc, 0.5f);

    for (int f = 0; f < NFREQ; ++f) {
        filt_kernel<<<NROWS / 64, 64, 0, stream>>>(xf, filters, imp, f, yext);
        gemm_h_kernel<<<gg, 256, 0, stream>>>(xf, w1p, b1p, Hbuf);
        accum_kernel<<<NROWS / 8, 256, 0, stream>>>(Hbuf, w2p, b2p, acc, 1.0f);
    }

    // y[-1] = model(zeros) folded in with weight 0.5, then * dx
    finalize_kernel<<<(NROWS + 255) / 256, 256, 0, stream>>>(b1p, w2p, b2p, acc, out);
}
